// PedestrianTrajectoryModel_87814901334189
// MI455X (gfx1250) — matrix operand order repaired and checked by reading, not compiled, not run on hardware
//
#include <hip/hip_runtime.h>
#include <hip/hip_bf16.h>
#include <math.h>

// ---------------------------------------------------------------------------
// PedestrianTrajectoryModel for MI455X (gfx1250, wave32, WMMA)
//
// Sizes: B=32 T=8 N=512 F=4 H=64 CO=12.  BT = B*T = 256.
// Heavy stage: per (b,t): softmax(mask(leaky_relu(a_dst_i + a_src_j))) @ Wx
//   -> 256 GEMMs of [512x512]@[512x64], done with v_wmma_f32_16x16x32_f16.
// Bandwidth floor: adj = 268 MB read once, coalesced.
// ---------------------------------------------------------------------------

typedef __attribute__((ext_vector_type(16))) _Float16 v16h;
typedef __attribute__((ext_vector_type(8)))  _Float16 half8;
typedef __attribute__((ext_vector_type(8)))  float    v8f;

#define BT   256
#define NN   512
#define HH   64
#define EPAD 516   // 512 + 4 dwords: bank-conflict-free column writes
#define PPAD 520   // 512 + 8 halfs: keeps 16B alignment of 8-half runs

// ---------------------------------------------------------------------------
// K1: Wx projection.  WxT16[bt][h][j] = (f16)(x[bt,j,:] . gat_W[:,h])
//     a_src[bt,j] = x . (gat_W @ att_src) ; a_dst likewise (exact refactor).
// ---------------------------------------------------------------------------
__global__ void __launch_bounds__(256)
k1_project(const float* __restrict__ x, const float* __restrict__ gat_W,
           const float* __restrict__ att_src, const float* __restrict__ att_dst,
           _Float16* __restrict__ WxT16,
           float* __restrict__ a_src, float* __restrict__ a_dst) {
  __shared__ float sW[4 * 64];
  __shared__ float sws[4], swd[4];
  const int tid = threadIdx.x;
  sW[tid] = gat_W[tid];                 // 256 threads == 256 weights
  __syncthreads();
  if (tid < 8) {
    const int f = tid & 3;
    const float* att = (tid < 4) ? att_src : att_dst;
    float s = 0.f;
    for (int h = 0; h < 64; ++h) s += sW[f * 64 + h] * att[h];
    if (tid < 4) sws[f] = s; else swd[f] = s;
  }
  __syncthreads();

  const int gid = blockIdx.x * 256 + tid;     // 0 .. BT*NN-1
  const int bt  = gid >> 9;
  const int n   = gid & 511;
  const float4 xv = *(const float4*)(x + (size_t)gid * 4);

  a_src[gid] = xv.x * sws[0] + xv.y * sws[1] + xv.z * sws[2] + xv.w * sws[3];
  a_dst[gid] = xv.x * swd[0] + xv.y * swd[1] + xv.z * swd[2] + xv.w * swd[3];

  const size_t base = (size_t)bt * HH * NN;
  #pragma unroll
  for (int h = 0; h < 64; ++h) {
    const float v = xv.x * sW[h] + xv.y * sW[64 + h] +
                    xv.z * sW[128 + h] + xv.w * sW[192 + h];
    WxT16[base + (size_t)h * NN + n] = (_Float16)v;   // coalesced over n
  }
}

// ---------------------------------------------------------------------------
// K2: attention softmax + P @ Wx via WMMA.
//   One block per (bt, 16-row i-tile). 128 threads = 4 waves.
//   mask[i,j] = (adj[bt,j,i] != 0) || (i == j);  e = leaky_relu(ad_i+as_j,0.2)
// ---------------------------------------------------------------------------
__global__ void __launch_bounds__(128)
k2_attn_wmma(const float* __restrict__ adj, const _Float16* __restrict__ WxT16,
             const float* __restrict__ a_src, const float* __restrict__ a_dst,
             const float* __restrict__ gat_bias, float* __restrict__ gat_out) {
  __shared__ float    sE[16][EPAD];
  __shared__ _Float16 sP[16][PPAD];
  __shared__ float    sAs[NN];
  __shared__ float    sAd[16];
  __shared__ float    sRed[16][8];
  __shared__ float    sM[16], sL[16];
  __shared__ float    sBias[64];

  const int tid = threadIdx.x;
  const int bt  = blockIdx.x >> 5;
  const int i0  = (blockIdx.x & 31) << 4;

  for (int j = tid; j < NN; j += 128) sAs[j] = a_src[(size_t)bt * NN + j];
  if (tid < 16) sAd[tid]  = a_dst[(size_t)bt * NN + i0 + tid];
  if (tid < 64) sBias[tid] = gat_bias[tid];
  __syncthreads();

  // ---- pass A: e tile (coalesced adj reads: lanes walk i, memory-contig) ---
  const float* adjbt = adj + (size_t)bt * NN * NN;
  for (int it = tid; it < 16 * NN; it += 128) {
    const int ii = it & 15;
    const int j  = it >> 4;
    const float av = adjbt[(size_t)j * NN + i0 + ii];
    const float v  = sAd[ii] + sAs[j];
    const float e  = v > 0.f ? v : 0.2f * v;
    const bool  mk = (av != 0.f) || (j == i0 + ii);
    sE[ii][j] = mk ? e : -1e9f;
  }
  __syncthreads();

  // ---- pass B: row max, exp, sum ----
  const int r = tid >> 3, s = tid & 7;
  float lm = -3.4e38f;
  for (int j = s; j < NN; j += 8) lm = fmaxf(lm, sE[r][j]);
  sRed[r][s] = lm;
  __syncthreads();
  if (s == 0) {
    float m = sRed[r][0];
    #pragma unroll
    for (int q = 1; q < 8; ++q) m = fmaxf(m, sRed[r][q]);
    sM[r] = m;
  }
  __syncthreads();
  const float mr = sM[r];
  float ls = 0.f;
  for (int j = s; j < NN; j += 8) {
    const float t = __expf(sE[r][j] - mr);
    sE[r][j] = t;
    ls += t;
  }
  sRed[r][s] = ls;
  __syncthreads();
  if (s == 0) {
    float t = 0.f;
    #pragma unroll
    for (int q = 0; q < 8; ++q) t += sRed[r][q];
    sL[r] = t;
  }
  __syncthreads();
  // normalize -> f16 P
  for (int it = tid; it < 16 * NN; it += 128) {
    const int row = it >> 9, col = it & 511;
    sP[row][col] = (_Float16)(sE[row][col] * (1.f / sL[row]));
  }
  __syncthreads();

  // ---- pass C: gat_out[i0..i0+15, :] = P @ Wx  (16x512 @ 512x64) ----
  // wave w owns h-tile [16w, 16w+16). 16 K-chunks of 32.
  const int w    = tid >> 5;
  const int lane = tid & 31;
  const int hb   = w << 4;
  const int m16  = lane & 15;   // A row / B col / D col within tile
  const int g    = lane >> 4;   // lane half-group

  v8f acc = {};
  const _Float16* Bbase = WxT16 + ((size_t)bt * HH + hb + m16) * NN;
  #pragma unroll 4
  for (int kc = 0; kc < 16; ++kc) {
    const int jb = kc << 5;
    // A (16x32 f16): lane row m holds K=16g..16g+15, contiguous (matches B)
    const half8* a0 = (const half8*)&sP[m16][jb + 16 * g];
    const half8* a1 = (const half8*)&sP[m16][jb + 16 * g + 8];
    // B (32x16 f16): lane n holds K=16g..16g+15, contiguous in WxT16
    const half8* b0 = (const half8*)(Bbase + jb + 16 * g);
    v16h Af, Bf;
    #pragma unroll
    for (int q = 0; q < 8; ++q) {
      Af[q]     = (*a0)[q];
      Af[q + 8] = (*a1)[q];
      Bf[q]     = b0[0][q];
      Bf[q + 8] = b0[1][q];
    }
    if (kc + 1 < 16)
      __builtin_prefetch((const void*)(Bbase + ((kc + 1) << 5) + 16 * g), 0, 3);
    acc = __builtin_amdgcn_wmma_f32_16x16x32_f16(
        /*neg_a=*/false, Af, /*neg_b=*/false, Bf,
        /*c_mod=*/(short)0, acc, /*reuse_a=*/false, /*reuse_b=*/false);
  }

  // D layout: VGPR rr -> row rr+8g, col = lane&15
  float* obase = gat_out + ((size_t)bt * NN + i0) * HH;
  #pragma unroll
  for (int rr = 0; rr < 8; ++rr) {
    const int mi = rr + 8 * g;
    const int h  = hb + m16;
    obase[(size_t)mi * HH + h] = acc[rr] + sBias[h];
  }
}

// ---------------------------------------------------------------------------
// K3: gates_x[row, 0..11] = gat_out[row, :64] @ W_ih^T + b_ih
//     (raw-view: GRU sequence m occupies 8 consecutive flat rows)
// ---------------------------------------------------------------------------
__global__ void __launch_bounds__(256)
k3_gates(const float* __restrict__ gat_out, const float* __restrict__ W_ih,
         const float* __restrict__ b_ih, float* __restrict__ gates_x) {
  __shared__ float sW[12 * 64];
  __shared__ float sB[12];
  const int tid = threadIdx.x;
  for (int q = tid; q < 768; q += 256) sW[q] = W_ih[q];
  if (tid < 12) sB[tid] = b_ih[tid];
  __syncthreads();

  const int row = blockIdx.x * 256 + tid;          // < BT*NN = 131072
  const float* gr = gat_out + (size_t)row * 64;
  float xr[64];
  #pragma unroll
  for (int q = 0; q < 16; ++q) {
    const float4 v = *(const float4*)(gr + q * 4);
    xr[q * 4 + 0] = v.x; xr[q * 4 + 1] = v.y;
    xr[q * 4 + 2] = v.z; xr[q * 4 + 3] = v.w;
  }
  float* go = gates_x + (size_t)row * 12;
  #pragma unroll
  for (int gG = 0; gG < 12; ++gG) {
    float acc = sB[gG];
    const float* wr = sW + gG * 64;
    #pragma unroll
    for (int h = 0; h < 64; ++h) acc += xr[h] * wr[h];
    go[gG] = acc;
  }
}

// ---------------------------------------------------------------------------
// K4: GRU scan, hidden=4, T=8.  One thread per sequence m (16384).
// ---------------------------------------------------------------------------
__global__ void __launch_bounds__(256)
k4_gru(const float* __restrict__ gates_x, const float* __restrict__ W_hh,
       const float* __restrict__ b_hh, float* __restrict__ gru_out) {
  __shared__ float sW[48];
  __shared__ float sB[12];
  const int tid = threadIdx.x;
  if (tid < 48) sW[tid] = W_hh[tid];
  if (tid < 12) sB[tid] = b_hh[tid];
  __syncthreads();

  const int m = blockIdx.x * 256 + tid;            // < 16384
  float h[4] = {0.f, 0.f, 0.f, 0.f};
  for (int t = 0; t < 8; ++t) {
    const float* gx = gates_x + ((size_t)m * 8 + t) * 12;
    float gh[12];
    #pragma unroll
    for (int gG = 0; gG < 12; ++gG) {
      float a = sB[gG];
      #pragma unroll
      for (int f = 0; f < 4; ++f) a += h[f] * sW[gG * 4 + f];
      gh[gG] = a;
    }
    float hn[4];
    #pragma unroll
    for (int d = 0; d < 4; ++d) {
      const float rg = 1.f / (1.f + __expf(-(gx[d]     + gh[d])));
      const float zg = 1.f / (1.f + __expf(-(gx[4 + d] + gh[4 + d])));
      const float nc = tanhf(gx[8 + d] + rg * gh[8 + d]);
      hn[d] = (1.f - zg) * nc + zg * h[d];
    }
    float* o = gru_out + (size_t)m * 32 + t * 4;
    #pragma unroll
    for (int d = 0; d < 4; ++d) { h[d] = hn[d]; o[d] = hn[d]; }
  }
}

// ---------------------------------------------------------------------------
// K5: Conv2d(8->12, 3x3, pad=1) over [B, T=8, N=512, 4] + Linear(4->2), fused.
//     conv input g[b,ci,rr,cc] = gru_out[(b*512+rr)*32 + ci*4 + cc]
// ---------------------------------------------------------------------------
__global__ void __launch_bounds__(256)
k5_conv_lin(const float* __restrict__ gru_out, const float* __restrict__ conv_W,
            const float* __restrict__ conv_b, const float* __restrict__ out_W,
            const float* __restrict__ out_b, float* __restrict__ out) {
  __shared__ float sW[12 * 8 * 9];
  __shared__ float sCb[12], sOW[8], sOb[2];
  const int tid = threadIdx.x;
  for (int q = tid; q < 864; q += 256) sW[q] = conv_W[q];
  if (tid < 12) sCb[tid] = conv_b[tid];
  if (tid < 8)  sOW[tid] = out_W[tid];
  if (tid < 2)  sOb[tid] = out_b[tid];
  __syncthreads();

  const int gid = blockIdx.x * 256 + tid;          // < 32*12*512 = 196608
  const int b   = gid / (12 * 512);
  const int rem = gid - b * 12 * 512;
  const int co  = rem >> 9;
  const int n2  = rem & 511;

  float y[4] = {sCb[co], sCb[co], sCb[co], sCb[co]};
  #pragma unroll
  for (int ci = 0; ci < 8; ++ci) {
    #pragma unroll
    for (int dh = -1; dh <= 1; ++dh) {
      const int rr = n2 + dh;
      if (rr < 0 || rr >= 512) continue;
      const float* gp = gru_out + ((size_t)b * 512 + rr) * 32 + ci * 4;
      const float gv0 = gp[0], gv1 = gp[1], gv2 = gp[2], gv3 = gp[3];
      const float gv[4] = {gv0, gv1, gv2, gv3};
      const float* wr = sW + ((co * 8 + ci) * 3 + (dh + 1)) * 3;
      #pragma unroll
      for (int c = 0; c < 4; ++c) {
        float acc = 0.f;
        #pragma unroll
        for (int dw = -1; dw <= 1; ++dw) {
          const int cc = c + dw;
          if (cc >= 0 && cc < 4) acc += gv[cc] * wr[dw + 1];
        }
        y[c] += acc;
      }
    }
  }
  float o0 = sOb[0], o1 = sOb[1];
  #pragma unroll
  for (int c = 0; c < 4; ++c) { o0 += y[c] * sOW[c]; o1 += y[c] * sOW[4 + c]; }
  out[(size_t)gid * 2 + 0] = o0;
  out[(size_t)gid * 2 + 1] = o1;
}

// ---------------------------------------------------------------------------
// Launch: workspace layout (MB-aligned, ~57 MB total):
//   [ 0,16)MB  WxT16   f16 [256][64][512]
//   [16,16.5)  a_src   f32 [256*512]
//   [16.5,17)  a_dst   f32 [256*512]
//   [17,49)    gat_out f32 [256*512*64]
//   [49,55)    gates_x f32 [131072*12]
//   [55,57)    gru_out f32 [16384*32]
// ---------------------------------------------------------------------------
extern "C" void kernel_launch(void* const* d_in, const int* in_sizes, int n_in,
                              void* d_out, int out_size, void* d_ws, size_t ws_size,
                              hipStream_t stream) {
  const float* x        = (const float*)d_in[0];
  const float* adj      = (const float*)d_in[1];
  const float* gat_W    = (const float*)d_in[2];
  const float* att_src  = (const float*)d_in[3];
  const float* att_dst  = (const float*)d_in[4];
  const float* gat_bias = (const float*)d_in[5];
  const float* gru_Wih  = (const float*)d_in[6];
  const float* gru_Whh  = (const float*)d_in[7];
  const float* gru_bih  = (const float*)d_in[8];
  const float* gru_bhh  = (const float*)d_in[9];
  const float* conv_W   = (const float*)d_in[10];
  const float* conv_b   = (const float*)d_in[11];
  const float* out_W    = (const float*)d_in[12];
  const float* out_b    = (const float*)d_in[13];
  float* out = (float*)d_out;

  char* ws = (char*)d_ws;
  _Float16* WxT16  = (_Float16*)(ws);
  float* a_src     = (float*)(ws + (size_t)(16u << 20));
  float* a_dst     = (float*)(ws + (size_t)(16u << 20) + (1u << 19));
  float* gat_out   = (float*)(ws + (size_t)(17u << 20));
  float* gates_x   = (float*)(ws + (size_t)(49u << 20));
  float* gru_out   = (float*)(ws + (size_t)(55u << 20));

  k1_project<<<512, 256, 0, stream>>>(x, gat_W, att_src, att_dst,
                                      WxT16, a_src, a_dst);
  k2_attn_wmma<<<BT * (NN / 16), 128, 0, stream>>>(adj, WxT16, a_src, a_dst,
                                                   gat_bias, gat_out);
  k3_gates<<<512, 256, 0, stream>>>(gat_out, gru_Wih, gru_bih, gates_x);
  k4_gru<<<64, 256, 0, stream>>>(gates_x, gru_Whh, gru_bhh, gru_out);
  k5_conv_lin<<<768, 256, 0, stream>>>(gru_out, conv_W, conv_b,
                                       out_W, out_b, out);
}
